// VisionTransformer_36928128811709
// MI455X (gfx1250) — compile-verified, matrix operand order checked by reading, not run on hardware
//
#include <hip/hip_runtime.h>
#include <hip/hip_bf16.h>

// ---------------------------------------------------------------------------
// ViT-Base forward for MI455X (gfx1250, wave32, WMMA).
// All GEMMs run through v_wmma_f32_16x16x32_f16. Weights are repacked to f16
// [N,K] (B-transposed) each launch; the whole f16 weight set (~173MB) is
// L2-resident on the 192MB L2, so GEMMs are matrix-core bound, not HBM bound.
// Epilogue uses a wave-uniform full-tile fast path (no per-element exec
// masking) since all large GEMMs are padded to tile multiples.
// ---------------------------------------------------------------------------

typedef __attribute__((ext_vector_type(16))) _Float16 v16h;
typedef __attribute__((ext_vector_type(8)))  _Float16 v8h;
typedef __attribute__((ext_vector_type(8)))  float    v8f;

#define BB    32
#define DM    768
#define HH    12
#define HDD   64
#define LL    12
#define HIDN  3072
#define NPP   196     // 14x14 patches
#define SS    197     // tokens
#define SPAD  224     // padded token dim (multiple of 32)
#define MTOK  (BB * SPAD)   // 7168
#define NCLS  1000

enum { EPI_F32 = 0, EPI_BIAS_F32, EPI_GELU_H, EPI_QKV, EPI_CTX, EPI_EMBED };

struct GemmP {
    const _Float16* A;      // [M,K] row-major, f16
    const _Float16* B;      // [N,K] row-major (i.e. B^T), f16
    int M, N, K, lda, ldb;
    long long strideA, strideB;   // per-batch (blockIdx.z) element strides
    float*     outF;
    _Float16*  outH;
    const float* bias;
    int ldo;
    long long strideO;
    int SP, H;
    _Float16 *q, *k, *v;    // QKV scatter targets
};

// ---- fragment load: ISA 16-bit A-matrix 16x32 layout ----------------------
// lane L: row = L%16, kbase = (L/16)*8; elems 0..7 -> K=kbase..kbase+7,
// elems 8..15 -> K=kbase+16..kbase+23. Two contiguous 16B chunks per lane.
__device__ __forceinline__ v16h load_frag(const _Float16* mat, int r0, int ld,
                                          int rmax, int k) {
    int lane = threadIdx.x & 31;
    int r = r0 + (lane & 15);
    r = r < rmax ? r : rmax;
    int kb = k + ((lane >> 4) << 3);
    const _Float16* p = mat + (long long)r * ld + kb;
    v8h lo = *reinterpret_cast<const v8h*>(p);
    v8h hi = *reinterpret_cast<const v8h*>(p + 16);
    v16h f;
#pragma unroll
    for (int i = 0; i < 8; ++i) { f[i] = lo[i]; f[i + 8] = hi[i]; }
    return f;
}

template <int EPI>
__device__ __forceinline__ void epi_elem(const GemmP& p, int z, int row,
                                         int col, float v) {
    if constexpr (EPI == EPI_F32) {
        p.outF[(long long)z * p.strideO + (long long)row * p.ldo + col] = v;
    } else if constexpr (EPI == EPI_BIAS_F32) {
        p.outF[(long long)row * p.ldo + col] = v + p.bias[col];
    } else if constexpr (EPI == EPI_GELU_H) {
        float t = v + p.bias[col];
        float g = 0.5f * t * (1.0f + erff(t * 0.70710678118654752f));
        p.outH[(long long)row * p.ldo + col] = (_Float16)g;
    } else if constexpr (EPI == EPI_QKV) {
        int b = row / p.SP, s = row % p.SP;
        int sect = col / DM, jj = col % DM;
        int h_ = jj >> 6, e = jj & 63;
        int bh = b * p.H + h_;
        _Float16 hv = (_Float16)v;
        if (sect == 0)      p.q[((long long)bh * p.SP + s) * HDD + e] = hv;
        else if (sect == 1) p.k[((long long)bh * p.SP + s) * HDD + e] = hv;
        else                p.v[((long long)bh * HDD + e) * p.SP + s] = hv;
    } else if constexpr (EPI == EPI_CTX) {
        int b = z / p.H, h_ = z % p.H;
        p.outH[((long long)(b * p.SP + row)) * DM + h_ * HDD + col] = (_Float16)v;
    } else if constexpr (EPI == EPI_EMBED) {
        int b = row / NPP, s = 1 + row % NPP;
        p.outF[((long long)(b * p.SP + s)) * DM + col] = v + p.bias[col];
    }
}

// ---- core GEMM: 128 threads = 4 waves (2x2); wave tile 32x64; block 64x128 -
template <int EPI>
__global__ __launch_bounds__(128) void gemm_wmma(GemmP p) {
    int z = blockIdx.z;
    const _Float16* A = p.A + (long long)z * p.strideA;
    const _Float16* B = p.B + (long long)z * p.strideB;
    int wave = threadIdx.x >> 5;
    int wm = wave & 1, wn = wave >> 1;
    int m0 = blockIdx.y * 64 + wm * 32;
    int n0 = blockIdx.x * 128 + wn * 64;

    v8f zero = {};
    v8f acc[2][4];
#pragma unroll
    for (int mi = 0; mi < 2; ++mi)
#pragma unroll
        for (int ni = 0; ni < 4; ++ni) acc[mi][ni] = zero;

#pragma unroll 2
    for (int k = 0; k < p.K; k += 32) {
        v16h a0 = load_frag(A, m0,      p.lda, p.M - 1, k);
        v16h a1 = load_frag(A, m0 + 16, p.lda, p.M - 1, k);
        v16h bf[4];
#pragma unroll
        for (int ni = 0; ni < 4; ++ni)
            bf[ni] = load_frag(B, n0 + ni * 16, p.ldb, p.N - 1, k);
#pragma unroll
        for (int ni = 0; ni < 4; ++ni) {
            acc[0][ni] = __builtin_amdgcn_wmma_f32_16x16x32_f16(
                false, a0, false, bf[ni], (short)0, acc[0][ni], false, false);
            acc[1][ni] = __builtin_amdgcn_wmma_f32_16x16x32_f16(
                false, a1, false, bf[ni], (short)0, acc[1][ni], false, false);
        }
    }

    int lane  = threadIdx.x & 31;
    int rbase = (lane >> 4) * 8;
    int cin   = lane & 15;
    // Wave-uniform full-tile test: avoids per-element exec masking on the
    // hot path (all big GEMMs are padded to tile multiples).
    if (m0 + 31 < p.M && n0 + 63 < p.N) {
#pragma unroll
        for (int mi = 0; mi < 2; ++mi)
#pragma unroll
            for (int ni = 0; ni < 4; ++ni)
#pragma unroll
                for (int i = 0; i < 8; ++i) {
                    int row = m0 + mi * 16 + rbase + i;
                    int col = n0 + ni * 16 + cin;
                    epi_elem<EPI>(p, z, row, col, acc[mi][ni][i]);
                }
    } else {
#pragma unroll
        for (int mi = 0; mi < 2; ++mi)
#pragma unroll
            for (int ni = 0; ni < 4; ++ni)
#pragma unroll
                for (int i = 0; i < 8; ++i) {
                    int row = m0 + mi * 16 + rbase + i;
                    int col = n0 + ni * 16 + cin;
                    if (row < p.M && col < p.N)
                        epi_elem<EPI>(p, z, row, col, acc[mi][ni][i]);
                }
    }
}

static inline void launch_gemm(int epi, const GemmP& p, int batch,
                               hipStream_t st) {
    dim3 g((unsigned)((p.N + 127) / 128), (unsigned)((p.M + 63) / 64),
           (unsigned)batch);
    dim3 b(128);
    switch (epi) {
        case EPI_F32:      gemm_wmma<EPI_F32><<<g, b, 0, st>>>(p); break;
        case EPI_BIAS_F32: gemm_wmma<EPI_BIAS_F32><<<g, b, 0, st>>>(p); break;
        case EPI_GELU_H:   gemm_wmma<EPI_GELU_H><<<g, b, 0, st>>>(p); break;
        case EPI_QKV:      gemm_wmma<EPI_QKV><<<g, b, 0, st>>>(p); break;
        case EPI_CTX:      gemm_wmma<EPI_CTX><<<g, b, 0, st>>>(p); break;
        case EPI_EMBED:    gemm_wmma<EPI_EMBED><<<g, b, 0, st>>>(p); break;
    }
}

// ---------------------------- helper kernels -------------------------------

__global__ void k_im2col(const float* __restrict__ x, _Float16* __restrict__ o) {
    int idx = blockIdx.x * 256 + threadIdx.x;
    if (idx >= BB * NPP * DM) return;
    int row = idx / DM, kk = idx % DM;
    int b = row / NPP, pi = row % NPP;
    int c = kk >> 8, rem = kk & 255, kh = rem >> 4, kw = rem & 15;
    int py = pi / 14, px = pi % 14;
    int xi = ((b * 3 + c) * 224 + py * 16 + kh) * 224 + px * 16 + kw;
    o[idx] = (_Float16)x[xi];
}

__global__ void k_cvt_f16(const float* __restrict__ in,
                          _Float16* __restrict__ out, int n) {
    int idx = blockIdx.x * 256 + threadIdx.x;
    if (idx < n) out[idx] = (_Float16)in[idx];
}

// out[(l*N+n)*K+k] = in[(l*K+k)*N+n]
__global__ void k_pack_tr(const float* __restrict__ in,
                          _Float16* __restrict__ out, long long total,
                          int K, int N) {
    long long idx = (long long)blockIdx.x * 256 + threadIdx.x;
    if (idx >= total) return;
    int k = (int)(idx % K);
    long long t = idx / K;
    int n = (int)(t % N);
    long long l = t / N;
    out[idx] = (_Float16)in[(l * K + k) * N + n];
}

// Wqkv[(l*2304+j)*768+d]; j<768:q, <1536:k, else v; jj=j%768 -> (h,e)
__global__ void k_pack_qkv(const float* __restrict__ wq,
                           const float* __restrict__ wk,
                           const float* __restrict__ wv,
                           _Float16* __restrict__ out) {
    long long idx = (long long)blockIdx.x * 256 + threadIdx.x;
    if (idx >= (long long)LL * 3 * DM * DM) return;
    int d = (int)(idx % DM);
    long long t = idx / DM;
    int j = (int)(t % (3 * DM));
    int l = (int)(t / (3 * DM));
    int sect = j / DM, jj = j % DM;
    int h_ = jj >> 6, e = jj & 63;
    const float* src = (sect == 0) ? wq : (sect == 1) ? wk : wv;
    out[idx] = (_Float16)src[(((long long)(l * HH + h_) * DM + d) * HDD + e)];
}

__global__ void k_finalize(float* __restrict__ h,
                           const float* __restrict__ cls_tok) {
    int idx = blockIdx.x * 256 + threadIdx.x;
    if (idx >= MTOK * DM) return;
    int row = idx / DM, d = idx % DM;
    int s = row % SPAD;
    if (s >= SS) { h[idx] = 0.0f; return; }
    int i = d >> 1;
    float den = __expf((float)(2 * i) * (9.210340371976184f / (float)DM));
    float ang = (float)s / den;
    float pev = (d & 1) ? __cosf(ang) : __sinf(ang);
    if (s == 0) h[idx] = cls_tok[d] + pev;
    else        h[idx] += pev;
}

__global__ void k_cvt_h(const float* __restrict__ h,
                        _Float16* __restrict__ hb) {
    int idx = blockIdx.x * 256 + threadIdx.x;
    if (idx >= MTOK * DM) return;
    int s = (idx / DM) % SPAD;
    hb[idx] = (s < SS) ? (_Float16)h[idx] : (_Float16)0.0f;
}

__device__ __forceinline__ float blk_sum(float v, float* red) {
    int tid = threadIdx.x;
    red[tid] = v; __syncthreads();
    for (int s = 128; s > 0; s >>= 1) {
        if (tid < s) red[tid] += red[tid + s];
        __syncthreads();
    }
    float r = red[0]; __syncthreads();
    return r;
}
__device__ __forceinline__ float blk_max(float v, float* red) {
    int tid = threadIdx.x;
    red[tid] = v; __syncthreads();
    for (int s = 128; s > 0; s >>= 1) {
        if (tid < s) red[tid] = fmaxf(red[tid], red[tid + s]);
        __syncthreads();
    }
    float r = red[0]; __syncthreads();
    return r;
}

// h[row] += LN(x[row]) * sc + bi   (rows with s>=197 untouched)
__global__ void k_ln_add(float* __restrict__ h, const float* __restrict__ x,
                         const float* __restrict__ sc,
                         const float* __restrict__ bi) {
    __shared__ float red[256];
    int row = blockIdx.x;
    if (row % SPAD >= SS) return;
    const float* xr = x + (long long)row * DM;
    float ls = 0.f;
    for (int d = threadIdx.x; d < DM; d += 256) ls += xr[d];
    float mean = blk_sum(ls, red) / (float)DM;
    float lv = 0.f;
    for (int d = threadIdx.x; d < DM; d += 256) {
        float t = xr[d] - mean; lv += t * t;
    }
    float var = blk_sum(lv, red) / (float)DM;
    float rstd = rsqrtf(var + 1e-5f);
    float* hr = h + (long long)row * DM;
    for (int d = threadIdx.x; d < DM; d += 256)
        hr[d] += (xr[d] - mean) * rstd * sc[d] + bi[d];
}

// softmax over t (197 real cols), scale 1/sqrt(64); pad rows/cols -> 0
__global__ void k_softmax_att(const float* __restrict__ scores,
                              _Float16* __restrict__ att) {
    __shared__ float red[256];
    int row = blockIdx.x;               // over B*H*SPAD
    int s = row % SPAD;
    long long base = (long long)row * SPAD;
    int tid = threadIdx.x;
    if (s >= SS) {
        if (tid < SPAD) att[base + tid] = (_Float16)0.0f;
        return;
    }
    float val = (tid < SS) ? scores[base + tid] * 0.125f : -1e30f;
    float mx = blk_max(val, red);
    float e = (tid < SS) ? __expf(val - mx) : 0.0f;
    float sum = blk_sum(e, red);
    if (tid < SPAD) att[base + tid] = (_Float16)((tid < SS) ? e / sum : 0.0f);
}

__global__ void k_extract_cls(const float* __restrict__ h,
                              _Float16* __restrict__ out) {
    int idx = blockIdx.x * 256 + threadIdx.x;
    if (idx >= BB * DM) return;
    int b = idx / DM, d = idx % DM;
    out[idx] = (_Float16)h[((long long)b * SPAD) * DM + d];  // s==0 row
}

__global__ void k_softmax_cls(const float* __restrict__ logits,
                              float* __restrict__ out) {
    __shared__ float red[256];
    int row = blockIdx.x;  // 32
    const float* lg = logits + (long long)row * NCLS;
    int tid = threadIdx.x;
    float lm = -1e30f;
    for (int c = tid; c < NCLS; c += 256) lm = fmaxf(lm, lg[c]);
    float mx = blk_max(lm, red);
    float ls = 0.f;
    for (int c = tid; c < NCLS; c += 256) ls += __expf(lg[c] - mx);
    float sum = blk_sum(ls, red);
    for (int c = tid; c < NCLS; c += 256)
        out[(long long)row * NCLS + c] = __expf(lg[c] - mx) / sum;
}

// ------------------------------- driver ------------------------------------

static inline unsigned nblk(long long n, int t) {
    return (unsigned)((n + t - 1) / t);
}

extern "C" void kernel_launch(void* const* d_in, const int* in_sizes, int n_in,
                              void* d_out, int out_size, void* d_ws,
                              size_t ws_size, hipStream_t stream) {
    const float* x      = (const float*)d_in[0];
    const float* conv_w = (const float*)d_in[1];
    const float* conv_b = (const float*)d_in[2];
    const float* cls_t  = (const float*)d_in[3];
    const float* wq     = (const float*)d_in[4];
    const float* wk     = (const float*)d_in[5];
    const float* wv     = (const float*)d_in[6];
    const float* wo     = (const float*)d_in[7];
    const float* ln1_s  = (const float*)d_in[8];
    const float* ln1_b  = (const float*)d_in[9];
    const float* w1     = (const float*)d_in[10];
    const float* b1     = (const float*)d_in[11];
    const float* w2     = (const float*)d_in[12];
    const float* b2     = (const float*)d_in[13];
    const float* ln2_s  = (const float*)d_in[14];
    const float* ln2_b  = (const float*)d_in[15];
    const float* clf_w  = (const float*)d_in[16];
    const float* clf_b  = (const float*)d_in[17];

    char* wsp = (char*)d_ws;
    auto alloc = [&](size_t bytes) -> void* {
        void* p = (void*)wsp;
        wsp += (bytes + 255) & ~(size_t)255;
        return p;
    };
    // scratch (~441 MB total)
    _Float16* Apat = (_Float16*)alloc((size_t)BB * NPP * DM * 2);
    _Float16* Wc   = (_Float16*)alloc((size_t)DM * DM * 2);
    _Float16* Wqkv = (_Float16*)alloc((size_t)LL * 3 * DM * DM * 2);
    _Float16* Wo   = (_Float16*)alloc((size_t)LL * DM * DM * 2);
    _Float16* W1   = (_Float16*)alloc((size_t)LL * HIDN * DM * 2);
    _Float16* W2   = (_Float16*)alloc((size_t)LL * DM * HIDN * 2);
    _Float16* Wclf = (_Float16*)alloc((size_t)NCLS * DM * 2);
    float*    h    = (float*)alloc((size_t)MTOK * DM * 4);
    _Float16* hbf  = (_Float16*)alloc((size_t)MTOK * DM * 2);
    _Float16* qb   = (_Float16*)alloc((size_t)BB * HH * SPAD * HDD * 2);
    _Float16* kb   = (_Float16*)alloc((size_t)BB * HH * SPAD * HDD * 2);
    _Float16* vb   = (_Float16*)alloc((size_t)BB * HH * HDD * SPAD * 2);
    float*    sco  = (float*)alloc((size_t)BB * HH * SPAD * SPAD * 4);
    _Float16* att  = (_Float16*)alloc((size_t)BB * HH * SPAD * SPAD * 2);
    _Float16* ctx  = (_Float16*)alloc((size_t)MTOK * DM * 2);
    float*    tmp  = (float*)alloc((size_t)MTOK * DM * 4);
    _Float16* ffn1 = (_Float16*)alloc((size_t)MTOK * HIDN * 2);
    _Float16* Acls = (_Float16*)alloc((size_t)BB * DM * 2);
    float*    lgts = (float*)alloc((size_t)BB * NCLS * 4);

    // ---- weight/activation packing (f32 -> f16, B pre-transposed) ----
    k_im2col<<<nblk((long long)BB * NPP * DM, 256), 256, 0, stream>>>(x, Apat);
    k_cvt_f16<<<nblk(DM * DM, 256), 256, 0, stream>>>(conv_w, Wc, DM * DM);
    k_pack_qkv<<<nblk((long long)LL * 3 * DM * DM, 256), 256, 0, stream>>>(
        wq, wk, wv, Wqkv);
    k_pack_tr<<<nblk((long long)LL * DM * DM, 256), 256, 0, stream>>>(
        wo, Wo, (long long)LL * DM * DM, DM, DM);
    k_pack_tr<<<nblk((long long)LL * DM * HIDN, 256), 256, 0, stream>>>(
        w1, W1, (long long)LL * DM * HIDN, DM, HIDN);
    k_pack_tr<<<nblk((long long)LL * HIDN * DM, 256), 256, 0, stream>>>(
        w2, W2, (long long)LL * HIDN * DM, HIDN, DM);
    k_pack_tr<<<nblk((long long)DM * NCLS, 256), 256, 0, stream>>>(
        clf_w, Wclf, (long long)DM * NCLS, DM, NCLS);

    // ---- patch embedding ----
    {
        GemmP p{};
        p.A = Apat; p.B = Wc;
        p.M = BB * NPP; p.N = DM; p.K = DM; p.lda = DM; p.ldb = DM;
        p.outF = h; p.bias = conv_b; p.SP = SPAD;
        launch_gemm(EPI_EMBED, p, 1, stream);
    }
    k_finalize<<<nblk((long long)MTOK * DM, 256), 256, 0, stream>>>(h, cls_t);

    // ---- encoder layers ----
    for (int l = 0; l < LL; ++l) {
        k_cvt_h<<<nblk((long long)MTOK * DM, 256), 256, 0, stream>>>(h, hbf);

        GemmP pq{};
        pq.A = hbf; pq.B = Wqkv + (size_t)l * 3 * DM * DM;
        pq.M = MTOK; pq.N = 3 * DM; pq.K = DM; pq.lda = DM; pq.ldb = DM;
        pq.SP = SPAD; pq.H = HH; pq.q = qb; pq.k = kb; pq.v = vb;
        launch_gemm(EPI_QKV, pq, 1, stream);

        GemmP ps{};
        ps.A = qb; ps.B = kb;
        ps.M = SPAD; ps.N = SPAD; ps.K = HDD; ps.lda = HDD; ps.ldb = HDD;
        ps.strideA = (long long)SPAD * HDD; ps.strideB = (long long)SPAD * HDD;
        ps.outF = sco; ps.ldo = SPAD; ps.strideO = (long long)SPAD * SPAD;
        launch_gemm(EPI_F32, ps, BB * HH, stream);

        k_softmax_att<<<BB * HH * SPAD, 256, 0, stream>>>(sco, att);

        GemmP pc{};
        pc.A = att; pc.B = vb;
        pc.M = SPAD; pc.N = HDD; pc.K = SPAD; pc.lda = SPAD; pc.ldb = SPAD;
        pc.strideA = (long long)SPAD * SPAD; pc.strideB = (long long)HDD * SPAD;
        pc.outH = ctx; pc.SP = SPAD; pc.H = HH;
        launch_gemm(EPI_CTX, pc, BB * HH, stream);

        GemmP po{};
        po.A = ctx; po.B = Wo + (size_t)l * DM * DM;
        po.M = MTOK; po.N = DM; po.K = DM; po.lda = DM; po.ldb = DM;
        po.outF = tmp; po.ldo = DM;
        launch_gemm(EPI_F32, po, 1, stream);

        k_ln_add<<<MTOK, 256, 0, stream>>>(h, tmp, ln1_s + l * DM,
                                           ln1_b + l * DM);

        k_cvt_h<<<nblk((long long)MTOK * DM, 256), 256, 0, stream>>>(h, hbf);

        GemmP p1{};
        p1.A = hbf; p1.B = W1 + (size_t)l * HIDN * DM;
        p1.M = MTOK; p1.N = HIDN; p1.K = DM; p1.lda = DM; p1.ldb = DM;
        p1.outH = ffn1; p1.ldo = HIDN; p1.bias = b1 + l * HIDN;
        launch_gemm(EPI_GELU_H, p1, 1, stream);

        GemmP p2{};
        p2.A = ffn1; p2.B = W2 + (size_t)l * DM * HIDN;
        p2.M = MTOK; p2.N = DM; p2.K = HIDN; p2.lda = HIDN; p2.ldb = HIDN;
        p2.outF = tmp; p2.ldo = DM; p2.bias = b2 + l * DM;
        launch_gemm(EPI_BIAS_F32, p2, 1, stream);

        k_ln_add<<<MTOK, 256, 0, stream>>>(h, tmp, ln2_s + l * DM,
                                           ln2_b + l * DM);
    }

    // ---- classifier head ----
    k_extract_cls<<<nblk(BB * DM, 256), 256, 0, stream>>>(h, Acls);
    {
        GemmP p{};
        p.A = Acls; p.B = Wclf;
        p.M = BB; p.N = NCLS; p.K = DM; p.lda = DM; p.ldb = DM;
        p.outF = lgts; p.ldo = NCLS; p.bias = clf_b;
        launch_gemm(EPI_BIAS_F32, p, 1, stream);
    }
    k_softmax_cls<<<BB, 256, 0, stream>>>(lgts, (float*)d_out);
}
